// Conv2dTransposeNormalizedLRStyleGAN2_52226802319410
// MI455X (gfx1250) — matrix operand order repaired and checked by reading, not compiled, not run on hardware
//
#include <hip/hip_runtime.h>
#include <stdint.h>

typedef _Float16 half8  __attribute__((ext_vector_type(8)));
typedef _Float16 half16 __attribute__((ext_vector_type(16)));
typedef float    float8 __attribute__((ext_vector_type(8)));

#define IN_C   512
#define OUT_C  512
#define BATCH  16
#define HH     32
#define WW     32

// HE = sqrt(2)/sqrt(512*9) = sqrt(1/2304) = 1/48 exactly
#define HE_F   (1.0f / 48.0f)
#define GAIN_F 1.4142135623730951f
#define EPS_F  1e-8f

#define WMMA_F16(a, b, c) \
  __builtin_amdgcn_wmma_f32_16x16x32_f16(false, (a), false, (b), (short)0, (c), false, false)

// ---------------------------------------------------------------------------
// Pass A: w2[ic*512+oc] = sum over 9 taps of weight[ic,oc,:,:]^2
// ---------------------------------------------------------------------------
__global__ __launch_bounds__(256) void w2sum_kernel(
    const float* __restrict__ weight, float* __restrict__ w2) {
  int idx = blockIdx.x * 256 + threadIdx.x;     // idx = ic*512 + oc
  const float* p = weight + (size_t)idx * 9;
  float s = 0.f;
#pragma unroll
  for (int j = 0; j < 9; ++j) { float v = p[j]; s += v * v; }
  w2[idx] = s;
}

// ---------------------------------------------------------------------------
// Pass B: rscale[b*512+oc] = GAIN / sqrt(HE^2 * sum_ic style^2 * w2 + eps)
// ---------------------------------------------------------------------------
__global__ __launch_bounds__(256) void scale_kernel(
    const float* __restrict__ style, const float* __restrict__ w2,
    float* __restrict__ rscale) {
  int idx = blockIdx.x * 256 + threadIdx.x;     // idx = b*512 + oc
  int b  = idx >> 9;
  int oc = idx & 511;
  float s = 0.f;
  for (int ic = 0; ic < IN_C; ++ic) {
    float st = style[b * IN_C + ic];
    s += st * st * w2[ic * OUT_C + oc];
  }
  rscale[idx] = GAIN_F * rsqrtf(HE_F * HE_F * s + EPS_F);
}

// ---------------------------------------------------------------------------
// Pass C: wh[(t*512+oc)*512+ic] = (f16) weight[ic, oc, tap 8-t]  (flip+T)
// ---------------------------------------------------------------------------
__global__ __launch_bounds__(256) void wprep_kernel(
    const float* __restrict__ weight, _Float16* __restrict__ wh) {
  int idx  = blockIdx.x * 256 + threadIdx.x;
  int ic   = idx & 511;
  int rest = idx >> 9;
  int oc   = rest & 511;
  int t    = rest >> 9;
  wh[idx] = (_Float16)weight[((size_t)ic * OUT_C + oc) * 9 + (8 - t)];
}

// ---------------------------------------------------------------------------
// Pass D: xs = (f16)(inp * style * HE)   (pre-modulated input, same layout)
// ---------------------------------------------------------------------------
__global__ __launch_bounds__(256) void xmod_kernel(
    const float* __restrict__ inp, const float* __restrict__ style,
    _Float16* __restrict__ xs) {
  int idx = blockIdx.x * 256 + threadIdx.x;     // [b][ic][h][w]
  int bic = idx >> 10;                          // b*512 + ic
  xs[idx] = (_Float16)(inp[idx] * style[bic] * HE_F);
}

// ---------------------------------------------------------------------------
// Pass E: implicit-GEMM 3x3 conv, v_wmma_f32_16x16x32_f16, double-buffered.
// Block = 256 threads = 8 waves; tile = 128 oc x 64 px (2 output rows).
// Wave = 2x2 fragments (32 oc x 32 px).
// Slab i+1 weights DMA'd via global_load_async_to_lds_b128 and input halo
// prefetched into registers while slab i runs its 36 WMMAs.
// ---------------------------------------------------------------------------
__global__ __launch_bounds__(256) void conv_wmma_kernel(
    const _Float16* __restrict__ xs, const _Float16* __restrict__ wh,
    const float* __restrict__ bias, const float* __restrict__ rscale,
    float* __restrict__ out) {
  // sX: [row 0..3][cc 0..35][ic 0..31]  (cc = col+1 halo, ic contiguous)
  __shared__ __align__(32) _Float16 sX[2][4 * 36 * 32];
  // sW: [tap 0..8][oc 0..127][ic 0..31]
  __shared__ __align__(32) _Float16 sW[2][9 * 128 * 32];

  const int tid  = threadIdx.x;
  const int b    = blockIdx.z;
  const int h0   = blockIdx.y * 2;
  const int oc0  = blockIdx.x * 128;

  const int wave = tid >> 5;
  const int lane = tid & 31;
  const int wm   = wave >> 1;        // 0..3 : 32-oc band
  const int wn   = wave & 1;         // 0..1 : output row within pair
  const int l16  = lane & 15;
  const int hi   = lane >> 4;        // lane half selects K sub-range

  float8 acc00 = {}, acc01 = {}, acc10 = {}, acc11 = {};

  const uint32_t ldsW0 = (uint32_t)(uintptr_t)&sW[0][0];
  const uint32_t ldsW1 = (uint32_t)(uintptr_t)&sW[1][0];

  // issue 18 async 16B lane-copies of one weight slab into LDS buffer
  auto issueW = [&](int ic0, uint32_t ldsbase) {
#pragma unroll 1
    for (int idx = tid; idx < 9 * 128 * 4; idx += 256) {
      int ic8  = idx & 3;
      int rest = idx >> 2;
      int oc   = rest & 127;
      int t    = rest >> 7;
      const _Float16* src =
          wh + ((size_t)(t * OUT_C + oc0 + oc) * IN_C + ic0 + ic8 * 8);
      uint32_t dst = ldsbase + 2u * (uint32_t)((t * 128 + oc) * 32 + ic8 * 8);
      asm volatile("global_load_async_to_lds_b128 %0, %1, off"
                   :: "v"(dst), "v"(src) : "memory");
    }
  };

  // predicated zero-padded halo prefetch into registers (18 f16 / thread)
  _Float16 xreg[18];
  auto gloadX = [&](int ic0) {
#pragma unroll
    for (int j = 0; j < 18; ++j) {
      int idx  = tid + j * 256;
      int ic   = idx & 31;
      int rest = idx >> 5;
      int cc   = rest % 36;
      int r    = rest / 36;
      int col  = cc - 1;
      int hr   = h0 + r - 1;
      _Float16 v = (_Float16)0.f;
      if (cc < 34 && (unsigned)col < (unsigned)WW && (unsigned)hr < (unsigned)HH)
        v = xs[((b * IN_C + ic0 + ic) * HH + hr) * WW + col];
      xreg[j] = v;
    }
  };
  auto dswriteX = [&](_Float16* sx) {
#pragma unroll
    for (int j = 0; j < 18; ++j) {
      int idx  = tid + j * 256;
      int ic   = idx & 31;
      int rest = idx >> 5;
      int cc   = rest % 36;
      int r    = rest / 36;
      sx[(r * 36 + cc) * 32 + ic] = xreg[j];
    }
  };

  // ---- prologue: slab 0 in flight ----
  issueW(0, ldsW0);
  gloadX(0);

  for (int icb = 0; icb < IN_C / 32; ++icb) {
    const int cur = icb & 1;
    const int nxt = cur ^ 1;

    __syncthreads();   // (a) all waves done computing slab icb-1 (buffers[nxt])

    dswriteX(&sX[cur][0]);                 // commit slab icb's halo to LDS
    if (icb + 1 < IN_C / 32) {
      gloadX((icb + 1) * 32);              // prefetch next halo into regs
      issueW((icb + 1) * 32, cur ? ldsW0 : ldsW1);   // DMA next weights
      asm volatile("s_wait_asynccnt 0x12" ::: "memory");  // slab icb done
    } else {
      asm volatile("s_wait_asynccnt 0x0" ::: "memory");
    }

    __syncthreads();   // (b) slab icb fully resident in buffers[cur]

    const _Float16* cW = &sW[cur][0];
    const _Float16* cX = &sX[cur][0];

#pragma unroll
    for (int t = 0; t < 9; ++t) {
      const int kh = t / 3, kw = t % 3;

      const int m0 = wm * 32 + l16;
      const half8 a0lo = *(const half8*)&cW[(t * 128 + m0) * 32 + hi * 16];
      const half8 a0hi = *(const half8*)&cW[(t * 128 + m0) * 32 + hi * 16 + 8];
      half16 a0 = __builtin_shufflevector(a0lo, a0hi,
                    0, 1, 2, 3, 4, 5, 6, 7, 8, 9, 10, 11, 12, 13, 14, 15);
      const int m1 = m0 + 16;
      const half8 a1lo = *(const half8*)&cW[(t * 128 + m1) * 32 + hi * 16];
      const half8 a1hi = *(const half8*)&cW[(t * 128 + m1) * 32 + hi * 16 + 8];
      half16 a1 = __builtin_shufflevector(a1lo, a1hi,
                    0, 1, 2, 3, 4, 5, 6, 7, 8, 9, 10, 11, 12, 13, 14, 15);

      const int rr = wn + kh;                 // input row for this wave/tap
      half16 b0 = *(const half16*)&cX[(rr * 36 + (l16 + kw)) * 32 + hi * 16];
      half16 b1 = *(const half16*)&cX[(rr * 36 + (16 + l16 + kw)) * 32 + hi * 16];

      acc00 = WMMA_F16(a0, b0, acc00);
      acc01 = WMMA_F16(a0, b1, acc01);
      acc10 = WMMA_F16(a1, b0, acc10);
      acc11 = WMMA_F16(a1, b1, acc11);
    }
  }

  // ---- epilogue: demodulate + bias + gain ----
  const int hrow = h0 + wn;
#pragma unroll
  for (int r = 0; r < 8; ++r) {
    int oc_a = oc0 + wm * 32 + hi * 8 + r;    // C layout: VGPR r -> M = r + 8*hi
    int oc_b = oc_a + 16;
    float sa = rscale[b * OUT_C + oc_a];
    float sb = rscale[b * OUT_C + oc_b];
    float ba = bias[oc_a] * GAIN_F;
    float bb = bias[oc_b] * GAIN_F;
    size_t ra = ((size_t)(b * OUT_C + oc_a) * HH + hrow) * WW;
    size_t rb = ((size_t)(b * OUT_C + oc_b) * HH + hrow) * WW;
    out[ra + l16]      = acc00[r] * sa + ba;
    out[ra + 16 + l16] = acc01[r] * sa + ba;
    out[rb + l16]      = acc10[r] * sb + bb;
    out[rb + 16 + l16] = acc11[r] * sb + bb;
  }
}

// ---------------------------------------------------------------------------
extern "C" void kernel_launch(void* const* d_in, const int* in_sizes, int n_in,
                              void* d_out, int out_size, void* d_ws, size_t ws_size,
                              hipStream_t stream) {
  const float* inp    = (const float*)d_in[0];
  const float* style  = (const float*)d_in[1];
  const float* weight = (const float*)d_in[2];
  const float* bias   = (const float*)d_in[3];
  float* out = (float*)d_out;

  // workspace layout
  float*    w2     = (float*)d_ws;                        // 1 MB
  float*    rscale = w2 + IN_C * OUT_C;                   // 32 KB
  _Float16* wh     = (_Float16*)(rscale + BATCH * OUT_C); // 4.5 MB
  _Float16* xs     = wh + 9 * OUT_C * IN_C;               // 16 MB

  w2sum_kernel<<<(IN_C * OUT_C) / 256, 256, 0, stream>>>(weight, w2);
  scale_kernel<<<(BATCH * OUT_C) / 256, 256, 0, stream>>>(style, w2, rscale);
  wprep_kernel<<<(9 * OUT_C * IN_C) / 256, 256, 0, stream>>>(weight, wh);
  xmod_kernel<<<(BATCH * IN_C * HH * WW) / 256, 256, 0, stream>>>(inp, style, xs);
  conv_wmma_kernel<<<dim3(OUT_C / 128, HH / 2, BATCH), 256, 0, stream>>>(
      xs, wh, bias, rscale, out);
}
